// network_model_77833397338242
// MI455X (gfx1250) — matrix operand order repaired and checked by reading, not compiled, not run on hardware
//
#include <hip/hip_runtime.h>
#include <hip/hip_bf16.h>

// ---------------- constants ----------------
#define BATCH   32768
#define NSEG    8
#define FDIM    3
#define HDIM    256
#define H4      1024
#define K2      512
#define BNH     (BATCH * NSEG * HDIM)   // 67,108,864 floats per tensor

typedef __attribute__((ext_vector_type(16))) __bf16 v16bf;
typedef __attribute__((ext_vector_type(8)))  float  v8f;

union Frag { unsigned int u[8]; uint4 q[2]; v16bf v; };

__device__ __forceinline__ unsigned short f2bf(float f) {
    unsigned u = __float_as_uint(f);
    u += 0x7FFFu + ((u >> 16) & 1u);          // round-to-nearest-even
    return (unsigned short)(u >> 16);
}
__device__ __forceinline__ unsigned int pack_bf(float a, float b) {
    unsigned ua = __float_as_uint(a); ua += 0x7FFFu + ((ua >> 16) & 1u);
    unsigned ub = __float_as_uint(b); ub += 0x7FFFu + ((ub >> 16) & 1u);
    return (ua >> 16) | (ub & 0xFFFF0000u);
}
__device__ __forceinline__ float sigf(float x) { return 1.0f / (1.0f + __expf(-x)); }

// ---------------- zero the output ----------------
__global__ void k_zero(float4* out, long long n4) {
    long long i = (long long)blockIdx.x * blockDim.x + threadIdx.x;
    long long stride = (long long)gridDim.x * blockDim.x;
    float4 z = make_float4(0.f, 0.f, 0.f, 0.f);
    for (; i < n4; i += stride) out[i] = z;
}

// ---------------- f32 -> bf16 weight conversion ----------------
__global__ void k_cvt(const float* __restrict__ src, unsigned short* __restrict__ dst, int n) {
    int i = blockIdx.x * blockDim.x + threadIdx.x;
    if (i < n) dst[i] = f2bf(src[i]);
}

// catA[m, 0:512] = bf16(concat(h[m,sA,:], h[m,sB,:]))  -- one u32 (bf16 pair) per thread
__global__ void k_cvt_cat(const float* __restrict__ h, int sA, int sB,
                          unsigned int* __restrict__ dst) {
    int i = blockIdx.x * blockDim.x + threadIdx.x;        // pair index, B*256 total
    if (i >= BATCH * 256) return;
    int m = i >> 8, p = i & 255, k = 2 * p;
    const float* src = (k < HDIM)
        ? &h[((long long)m * NSEG + sA) * HDIM + k]
        : &h[((long long)m * NSEG + sB) * HDIM + (k - HDIM)];
    dst[(long long)m * 256 + p] = pack_bf(src[0], src[1]);
}

// half of sh[m, 0:512]: dst u32 row stride 256, halfOff = 0 or 128 (u32 units)
__global__ void k_cvt_half(const float* __restrict__ h, int slice, int halfOff,
                           unsigned int* __restrict__ dst) {
    int i = blockIdx.x * blockDim.x + threadIdx.x;        // pair index, B*128 total
    if (i >= BATCH * 128) return;
    int m = i >> 7, p = i & 127, k = 2 * p;
    const float* src = &h[((long long)m * NSEG + slice) * HDIM + k];
    dst[(long long)m * 256 + halfOff + p] = pack_bf(src[0], src[1]);
}

// ---------------- WMMA multi-tile core: A[M,K] bf16 row-major, W[N,K] bf16 row-major ----------
template<int KDIM, int NT, int TS>
__device__ __forceinline__ void wmma_mt(const unsigned short* __restrict__ A,
                                        const unsigned short* __restrict__ W,
                                        int tm, int tn0, int lane, v8f acc[NT]) {
    const int g = lane >> 4, ln = lane & 15;
    const long long arow = (long long)(tm * 16 + ln) * KDIM;
    long long brow[NT];
#pragma unroll
    for (int t = 0; t < NT; ++t)
        brow[t] = (long long)((tn0 + t * TS) * 16 + ln) * KDIM;

#pragma unroll
    for (int kb = 0; kb < KDIM; kb += 32) {
        Frag a;
        a.q[0] = *(const uint4*)&A[arow + kb + 8 * g];
        a.q[1] = *(const uint4*)&A[arow + kb + 8 * g + 16];
        if (kb + 32 < KDIM) __builtin_prefetch(&A[arow + kb + 32], 0, 1);
#pragma unroll
        for (int t = 0; t < NT; ++t) {
            Frag b;
            b.q[0] = *(const uint4*)&W[brow[t] + kb + 8 * g];
            b.q[1] = *(const uint4*)&W[brow[t] + kb + 8 * g + 16];
            acc[t] = __builtin_amdgcn_wmma_f32_16x16x32_bf16(false, a.v, false, b.v,
                                                             (short)0, acc[t], false, false);
        }
    }
}

// ---------------- GEMM1: h_after = catA @ Wr.T + br -> bf16 into half of sh ----------------
__global__ __launch_bounds__(256)
void k_gemm_wr(const unsigned short* __restrict__ catA, const unsigned short* __restrict__ Wbf,
               const float* __restrict__ bias, unsigned short* __restrict__ sh, int outOff) {
    const int lane = threadIdx.x & 31, wave = threadIdx.x >> 5;
    const int tile = blockIdx.x * 8 + wave;               // 8192 tiles
    const int tn0 = (tile & 3) * 4, tm = tile >> 2;
    const int g = lane >> 4, ln = lane & 15;

    v8f acc[4] = {{}, {}, {}, {}};
    wmma_mt<K2, 4, 1>(catA, Wbf, tm, tn0, lane, acc);

#pragma unroll
    for (int t = 0; t < 4; ++t)
#pragma unroll
        for (int r = 0; r < 8; ++r) {
            int m = tm * 16 + r + 8 * g;
            int n = (tn0 + t) * 16 + ln;
            sh[(long long)m * K2 + outOff + n] = f2bf(acc[t][r] + bias[n]);
        }
}

// ---------------- GEMM2: hh = h[:,idx] * sigmoid(sh @ Wsi.T + bsi) -> f32 + bf16 ----------------
__global__ __launch_bounds__(256)
void k_gemm_si(const unsigned short* __restrict__ sh, const unsigned short* __restrict__ Wbf,
               const float* __restrict__ bias, const float* __restrict__ h, int idx,
               float* __restrict__ hh_f32, unsigned short* __restrict__ hh_bf) {
    const int lane = threadIdx.x & 31, wave = threadIdx.x >> 5;
    const int tile = blockIdx.x * 8 + wave;
    const int tn0 = (tile & 3) * 4, tm = tile >> 2;
    const int g = lane >> 4, ln = lane & 15;

    v8f acc[4] = {{}, {}, {}, {}};
    wmma_mt<K2, 4, 1>(sh, Wbf, tm, tn0, lane, acc);

#pragma unroll
    for (int t = 0; t < 4; ++t)
#pragma unroll
        for (int r = 0; r < 8; ++r) {
            int m = tm * 16 + r + 8 * g;
            int n = (tn0 + t) * 16 + ln;
            float s  = sigf(acc[t][r] + bias[n]);
            float hv = h[((long long)m * NSEG + idx) * HDIM + n];
            float hh = hv * s;                            // hh == cc (h_s == c_s in reference)
            hh_f32[(long long)m * HDIM + n] = hh;
            hh_bf [(long long)m * HDIM + n] = f2bf(hh);
        }
}

// ---------------- GEMM3 + LSTM epilogue ----------------
__global__ __launch_bounds__(256)
void k_gemm_gates(const unsigned short* __restrict__ hh_bf, const float* __restrict__ hh_f32,
                  const unsigned short* __restrict__ Whhbf,
                  const float* __restrict__ bih, const float* __restrict__ bhh,
                  const float* __restrict__ Wih, const float* __restrict__ xin,
                  int idx, float* __restrict__ out) {
    const int lane = threadIdx.x & 31, wave = threadIdx.x >> 5;
    const int tile = blockIdx.x * 8 + wave;               // 32768 tiles
    const int tn0 = tile & 15, tm = tile >> 4;
    const int g = lane >> 4, ln = lane & 15;

    v8f acc[4] = {{}, {}, {}, {}};
    wmma_mt<HDIM, 4, 16>(hh_bf, Whhbf, tm, tn0, lane, acc);  // chunks at n, n+256, n+512, n+768

#pragma unroll
    for (int r = 0; r < 8; ++r) {
        int m = tm * 16 + r + 8 * g;
        int n = tn0 * 16 + ln;
        const float* x = &xin[((long long)m * NSEG + idx) * FDIM];
        float x0 = x[0], x1 = x[1], x2 = x[2];
        float gate[4];
#pragma unroll
        for (int c = 0; c < 4; ++c) {
            int ng = c * HDIM + n;
            gate[c] = acc[c][r] + bih[ng] + bhh[ng]
                    + x0 * Wih[(long long)ng * 3 + 0]
                    + x1 * Wih[(long long)ng * 3 + 1]
                    + x2 * Wih[(long long)ng * 3 + 2];
        }
        float cc = hh_f32[(long long)m * HDIM + n];
        float c2 = sigf(gate[1]) * cc + sigf(gate[0]) * tanhf(gate[2]);
        float h2 = sigf(gate[3]) * tanhf(c2);
        long long o = ((long long)m * NSEG + idx) * HDIM + n;
        out[o]       = h2;
        out[BNH + o] = c2;
    }
}

// ---------------- host ----------------
extern "C" void kernel_launch(void* const* d_in, const int* in_sizes, int n_in,
                              void* d_out, int out_size, void* d_ws, size_t ws_size,
                              hipStream_t stream) {
    (void)in_sizes; (void)n_in; (void)ws_size;
    const float* xin = (const float*)d_in[0];
    const float* h   = (const float*)d_in[1];
    // d_in[2] = c : unused (reference passes h as both h and c)
    float* out = (float*)d_out;

    // ws layout (bytes, all 256-aligned)
    char* ws = (char*)d_ws;
    unsigned short* WrBf  = (unsigned short*)(ws);                 // 3 * 256*512 bf16
    unsigned short* WsiBf = (unsigned short*)(ws + 786432);        // 3 * 256*512 bf16
    unsigned short* WhhBf = (unsigned short*)(ws + 1572864);       // 3 * 1024*256 bf16
    unsigned short* catA  = (unsigned short*)(ws + 3145728);       // B*512 bf16 (reused; aliases hh_bf)
    unsigned short* shA   = (unsigned short*)(ws + 36700160);      // B*512 bf16 (reused)
    float*          hhF   = (float*)(ws + 70254592);               // B*256 f32  (reused)
    unsigned short* hhBf  = catA;                                  // catA dead after GEMM1

    // zero entire output (slices {1,3,4} overwritten below)
    k_zero<<<8192, 256, 0, stream>>>((float4*)out, (long long)out_size / 4);

    // convert weights to bf16 (Wih/biases stay f32)
    for (int s = 0; s < 3; ++s) {
        const int p = 3 + 10 * s;
        const float* Whh = (const float*)d_in[p + 2];
        const float* Wsi = (const float*)d_in[p + 4];
        const float* Wr  = (const float*)d_in[p + 8];
        k_cvt<<<131072 / 256, 256, 0, stream>>>(Wr,  WrBf  + s * 131072, 131072);
        k_cvt<<<131072 / 256, 256, 0, stream>>>(Wsi, WsiBf + s * 131072, 131072);
        k_cvt<<<262144 / 256, 256, 0, stream>>>(Whh, WhhBf + s * 262144, 262144);
    }

    // segment tables
    const int segIdx[3]  = {1, 3, 4};
    const int catAsl[3]  = {3, 1, 1};
    const int catBsl[3]  = {4, 4, 3};
    // major/minor: sh = [Wr_result, h[:,before]] -> gemm1 writes low half, cvt fills high half
    // end:         sh = [h[:,5],    Wr_result  ] -> gemm1 writes high half, cvt fills low half
    const int g1Off[3]   = {0, 0, 256};       // bf16-element offset for GEMM1 output into sh
    const int cvSlice[3] = {0, 2, 5};         // h slice for the other half
    const int cvOff[3]   = {128, 128, 0};     // u32 offset of that half in sh row

    const int g12Blocks = (BATCH / 16) * 4 / 8;            // 8192 tiles / 8 waves = 1024
    const int g3Blocks  = (BATCH / 16) * (HDIM / 16) / 8;  // 32768 tiles / 8 waves = 4096

    for (int s = 0; s < 3; ++s) {
        const int p = 3 + 10 * s;
        const float* Wih = (const float*)d_in[p + 0];
        const float* bih = (const float*)d_in[p + 1];
        const float* bhh = (const float*)d_in[p + 3];
        const float* bsi = (const float*)d_in[p + 5];
        const float* br  = (const float*)d_in[p + 9];

        k_cvt_cat <<<BATCH * 256 / 256, 256, 0, stream>>>(h, catAsl[s], catBsl[s],
                                                          (unsigned int*)catA);
        k_cvt_half<<<BATCH * 128 / 256, 256, 0, stream>>>(h, cvSlice[s], cvOff[s],
                                                          (unsigned int*)shA);

        k_gemm_wr<<<g12Blocks, 256, 0, stream>>>(catA, WrBf + s * 131072, br, shA, g1Off[s]);

        k_gemm_si<<<g12Blocks, 256, 0, stream>>>(shA, WsiBf + s * 131072, bsi, h, segIdx[s],
                                                 hhF, hhBf);

        k_gemm_gates<<<g3Blocks, 256, 0, stream>>>(hhBf, hhF, WhhBf + s * 262144,
                                                   bih, bhh, Wih, xin, segIdx[s], out);
    }
}
